// MultiHeadAttn_22050362098062
// MI455X (gfx1250) — matrix operand order repaired and checked by reading, not compiled, not run on hardware
//
#include <hip/hip_runtime.h>

#define S_LEN   2048
#define DMODEL  1024
#define NHEAD   16
#define HDIM    64
#define BATCH   2
#define NPEMB   33
#define MROWS   (BATCH * S_LEN)   // 4096

typedef __attribute__((ext_vector_type(16))) __bf16        v16bf;
typedef __attribute__((ext_vector_type(8)))  float         v8f;
typedef __attribute__((ext_vector_type(4)))  unsigned int  u32x4;
typedef __attribute__((ext_vector_type(8)))  unsigned int  u32x8;
typedef unsigned short u16;

union FragU { v16bf v; u32x4 u[2]; };

__device__ __forceinline__ u16 f2bf(float x) {
  union { float f; unsigned int u; } c; c.f = x;
  unsigned int r = c.u + 0x7FFFu + ((c.u >> 16) & 1u);  // RNE
  return (u16)(r >> 16);
}
__device__ __forceinline__ float bf2f(u16 h) {
  union { unsigned int u; float f; } c; c.u = ((unsigned int)h) << 16; return c.f;
}
__device__ __forceinline__ void store_out(float* p, float v) { *p = v; }
__device__ __forceinline__ void store_out(u16* p, float v)   { *p = f2bf(v); }

__device__ __forceinline__ unsigned lds_off(const void* p) {
  // addrspace(3) -> generic keeps the LDS byte offset in the low 32 bits
  return (unsigned)(unsigned long long)p;
}

__device__ __forceinline__ v8f wmma_bf(v16bf a, v16bf b, v8f c) {
  return __builtin_amdgcn_wmma_f32_16x16x32_bf16(false, a, false, b, (short)0, c, false, false);
}

// A fragment (16x32 bf16), tile row-major [row][k].
// ISA 7.12.2: lanes 0-15: K = 0..7, 16..23 ; lanes 16-31: K = 8..15, 24..31.
__device__ __forceinline__ v16bf load_a_frag(const u16* tile, int ld, int row0, int k0, int lane) {
  int r  = row0 + (lane & 15);
  int kb = k0 + ((lane >> 4) << 3);
  const u16* p = tile + r * ld + kb;
  FragU f;
  f.u[0] = *(const u32x4*)(p);
  f.u[1] = *(const u32x4*)(p + 16);
  return f.v;
}
// B fragment (32x16 bf16), tile stored transposed [n][k] (column n's K contiguous).
// Lanes 0-15 hold K=0..7,16..23 of column n=lane; lanes 16-31 hold K=8..15,24..31.
__device__ __forceinline__ v16bf load_b_frag(const u16* tile, int ld, int col0, int k0, int lane) {
  int n  = col0 + (lane & 15);
  int kb = k0 + ((lane >> 4) << 3);
  const u16* p = tile + n * ld + kb;
  FragU f;
  f.u[0] = *(const u32x4*)(p);
  f.u[1] = *(const u32x4*)(p + 16);
  return f.v;
}

// ---------------------------------------------------------------------------
// TDM: DMA a 2D bf16 tile (tile_d0 x tile_d1 elems, row stride stride0 elems)
// from global to LDS, inserting `padDW` dwords of LDS padding after every
// `intervalDW` dwords (reproduces padded LDS rows). Issue once per block.
// D# layout per CDNA5 ISA.
// ---------------------------------------------------------------------------
__device__ __forceinline__ void tdm_load_2d_bf16(unsigned ldsOff, const void* gptr,
                                                 unsigned tensor_d0, unsigned tensor_d1,
                                                 unsigned stride0,
                                                 unsigned tile_d0, unsigned tile_d1,
                                                 unsigned pad_interval_code,
                                                 unsigned pad_amount_code) {
  unsigned long long ga = (unsigned long long)gptr;
  u32x4 g0;
  g0[0] = 1u;                                            // count = 1 (valid D#)
  g0[1] = ldsOff;                                        // lds_addr
  g0[2] = (unsigned)(ga & 0xffffffffu);                  // global_addr[31:0]
  g0[3] = (unsigned)((ga >> 32) & 0x1ffffffu) | (2u << 30);  // addr[56:32] | type=2
  u32x8 g1;
  g1[0] = (1u << 16)                                     // data_size = 2B
        | (1u << 20)                                     // pad_enable
        | (pad_interval_code << 22) | (pad_amount_code << 25);
  g1[1] = (tensor_d0 & 0xffffu) << 16;                   // tensor_dim0[15:0]
  g1[2] = ((tensor_d0 >> 16) & 0xffffu) | ((tensor_d1 & 0xffffu) << 16);
  g1[3] = ((tensor_d1 >> 16) & 0xffffu) | ((tile_d0 & 0xffffu) << 16);
  g1[4] = (tile_d1 & 0xffffu);                           // tile_dim1, tile_dim2=0
  g1[5] = stride0;                                       // tensor_dim0_stride[31:0]
  g1[6] = 0u;
  g1[7] = 0u;
  u32x4 gz = {0u, 0u, 0u, 0u};
  asm volatile("tensor_load_to_lds %0, %1, %2, %3"
               :: "s"(g0), "s"(g1), "s"(gz), "s"(gz) : "memory");
}

// ---------------------------------------------------------------------------
// GEMM: C[M,N] = A[M,K] @ B[K,N].  Block tile 128x64, K-step 32, 8 waves.
// ---------------------------------------------------------------------------
#define LDA 40   // 32 + 8 pad (bank stagger; rows stay 16B aligned)

// fp32 A: convert while staging (VALU path).
__device__ __forceinline__ void stage_a(const float* A, u16* As, int m0, int k0, int K, int tid) {
  #pragma unroll
  for (int i = 0; i < 16; ++i) {
    int idx = i * 256 + tid;
    int r = idx >> 5, c = idx & 31;
    As[r * LDA + c] = f2bf(A[(size_t)(m0 + r) * K + k0 + c]);
  }
}
// bf16 A: ASYNCcnt path — per-lane 16B global->LDS copies, no VGPR round trip.
__device__ __forceinline__ void stage_a(const u16* A, u16* As, int m0, int k0, int K, int tid) {
  unsigned base = lds_off(As);
  #pragma unroll
  for (int i = 0; i < 2; ++i) {
    int idx = i * 256 + tid;                 // 512 chunks of 16B (128 rows x 4)
    int r = idx >> 2, c = idx & 3;
    unsigned lds = base + (unsigned)(r * (LDA * 2) + c * 16);
    unsigned long long g = (unsigned long long)(A + (size_t)(m0 + r) * K + k0 + c * 8);
    asm volatile("global_load_async_to_lds_b128 %0, %1, off" :: "v"(lds), "v"(g) : "memory");
  }
}

template <typename TA, typename TO>
__global__ __launch_bounds__(256) void wmma_gemm_kernel(
    const TA* __restrict__ A, const float* __restrict__ Bm, TO* __restrict__ C,
    int M, int N, int K) {
  __shared__ __align__(16) u16 As[128 * LDA];
  __shared__ __align__(16) u16 Bts[64 * LDA];   // transposed [n][k]

  const int tid = threadIdx.x;
  const int wave = tid >> 5, lane = tid & 31;
  const int m0 = blockIdx.y * 128, n0 = blockIdx.x * 64;

  v8f acc[4] = {};

  for (int k0 = 0; k0 < K; k0 += 32) {
    stage_a(A, As, m0, k0, K, tid);
    #pragma unroll
    for (int i = 0; i < 8; ++i) {             // B: 32x64, convert + transpose
      int idx = i * 256 + tid;
      int kk = idx >> 6, n = idx & 63;
      Bts[n * LDA + kk] = f2bf(Bm[(size_t)(k0 + kk) * N + n0 + n]);
    }
    asm volatile("s_wait_asynccnt 0x0" ::: "memory");
    __syncthreads();

    v16bf a = load_a_frag(As, LDA, wave * 16, 0, lane);
    #pragma unroll
    for (int nt = 0; nt < 4; ++nt) {
      v16bf b = load_b_frag(Bts, LDA, nt * 16, 0, lane);
      acc[nt] = wmma_bf(a, b, acc[nt]);
    }
    __syncthreads();
  }

  const int hi = (lane >> 4) << 3, col = lane & 15;
  #pragma unroll
  for (int nt = 0; nt < 4; ++nt)
    #pragma unroll
    for (int vr = 0; vr < 8; ++vr) {
      int row = m0 + wave * 16 + vr + hi;
      store_out(C + (size_t)row * N + n0 + nt * 16 + col, acc[nt][vr]);
    }
}

// ---------------------------------------------------------------------------
// qe[(b*H+h)*S + q][j] = dot(q_head[b,h,q,:], rel_pemb[j,:])   (tiny)
// ---------------------------------------------------------------------------
__global__ __launch_bounds__(64) void qe_kernel(const u16* __restrict__ qb,
                                                const float* __restrict__ rel_pemb,
                                                float* __restrict__ qe) {
  __shared__ float qrow[HDIM];
  const int row = blockIdx.x;                  // (b*H + h)*S + q
  const int bh = row >> 11, q = row & (S_LEN - 1);
  const int b = bh >> 4, h = bh & 15;
  const int t = threadIdx.x;
  qrow[t] = bf2f(qb[((size_t)(b * S_LEN + q)) * DMODEL + h * HDIM + t]);
  __syncthreads();
  if (t < NPEMB) {
    float s = 0.f;
    #pragma unroll 8
    for (int d = 0; d < HDIM; ++d) s += qrow[d] * rel_pemb[t * HDIM + d];
    qe[(size_t)row * NPEMB + t] = s;
  }
}

// ---------------------------------------------------------------------------
// Fused flash attention. Block = one (b,h) x 64 q-rows. 8 waves:
//   wave w: mt = w&3 (16-row score slab), nh = w>>2 (half of N/d columns).
// Q and K tiles staged by the Tensor Data Mover (one instruction per tile,
// LDS row padding 128B+16B via D# pad fields); V transposed by VALU.
// ---------------------------------------------------------------------------
__global__ __launch_bounds__(256) void attn_kernel(const u16* __restrict__ qb,
                                                   const u16* __restrict__ kb,
                                                   const u16* __restrict__ vb,
                                                   const float* __restrict__ qe,
                                                   u16* __restrict__ ctxb) {
  constexpr int LD = 72;     // bf16 tiles: 144B rows = 128B data + 16B pad
  constexpr int LDS_S = 65;  // fp32 scores: +1 pad kills row-broadcast conflicts
  __shared__ __align__(16) u16 Qs [64 * LD];
  __shared__ __align__(16) u16 Ks [64 * LD];
  __shared__ __align__(16) u16 Vts[64 * LD];   // transposed [d][s]
  __shared__ __align__(16) u16 Ps [64 * LD];   // probabilities (bf16)
  __shared__ __align__(16) float Ss[64 * LDS_S];
  __shared__ float m_s[64], l_s[64], scale_s[64];

  const int tid = threadIdx.x, wave = tid >> 5, lane = tid & 31;
  const int bh = blockIdx.y, b = bh >> 4, h = bh & 15;
  const int q0 = blockIdx.x * 64;
  const size_t headBase = (size_t)b * S_LEN * DMODEL + (size_t)h * HDIM;

  // TDM-stage Q tile (64x64 bf16, row stride DMODEL elems). pad: every 32 DW
  // (code 4 = one 128B row) insert 4 DW (code 3 = 16B) -> LD=72 rows.
  if (wave == 0) {
    tdm_load_2d_bf16(lds_off(Qs), qb + headBase + (size_t)q0 * DMODEL,
                     HDIM, 64, DMODEL, HDIM, 64, 4u, 3u);
    __builtin_amdgcn_s_wait_tensorcnt(0);
  }
  if (tid < 64) { m_s[tid] = -1e30f; l_s[tid] = 0.f; }
  __syncthreads();

  const int mt = wave & 3, nh = wave >> 2;
  const v16bf aq0 = load_a_frag(Qs, LD, mt * 16, 0, lane);
  const v16bf aq1 = load_a_frag(Qs, LD, mt * 16, 32, lane);
  v8f o0 = {}, o1 = {};

  for (int st = 0; st < S_LEN / 64; ++st) {
    const int s0 = st * 64;
    if (wave == 0)                             // TDM-stage K tile
      tdm_load_2d_bf16(lds_off(Ks), kb + headBase + (size_t)s0 * DMODEL,
                       HDIM, 64, DMODEL, HDIM, 64, 4u, 3u);
    #pragma unroll
    for (int i = 0; i < 16; ++i) {             // stage V transposed (VALU)
      int idx = i * 256 + tid;
      int r = idx >> 6, d = idx & 63;
      Vts[d * LD + r] = vb[headBase + (size_t)(s0 + r) * DMODEL + d];
    }
    if (wave == 0) __builtin_amdgcn_s_wait_tensorcnt(0);
    __syncthreads();

    // scores: S = Q @ K^T  (B column n = K row s, K-dim = d, two 32-chunks)
    #pragma unroll
    for (int j = 0; j < 2; ++j) {
      const int nt = nh * 2 + j;
      v16bf b0 = load_b_frag(Ks, LD, nt * 16, 0, lane);
      v16bf b1 = load_b_frag(Ks, LD, nt * 16, 32, lane);
      v8f sc = {};
      sc = wmma_bf(aq0, b0, sc);
      sc = wmma_bf(aq1, b1, sc);
      const int hi = (lane >> 4) << 3, col = lane & 15;
      #pragma unroll
      for (int vr = 0; vr < 8; ++vr)
        Ss[(mt * 16 + vr + hi) * LDS_S + nt * 16 + col] = sc[vr];
    }
    __syncthreads();

    // online softmax over this s-tile (one thread per q row; wave-uniform branch)
    if (tid < 64) {
      const int qrow = q0 + tid;
      const float* qer = qe + ((size_t)bh * S_LEN + qrow) * NPEMB;
      float mloc = -1e30f;
      for (int c = 0; c < 64; ++c) {
        int rel = s0 + c - qrow; rel = rel < -16 ? -16 : (rel > 16 ? 16 : rel);
        float v = (Ss[tid * LDS_S + c] + qer[rel + 16]) * 0.125f;
        mloc = fmaxf(mloc, v);
      }
      const float mold = m_s[tid];
      const float mnew = fmaxf(mold, mloc);
      const float corr = __expf(mold - mnew);
      float sum = 0.f;
      for (int c = 0; c < 64; ++c) {
        int rel = s0 + c - qrow; rel = rel < -16 ? -16 : (rel > 16 ? 16 : rel);
        float v = (Ss[tid * LDS_S + c] + qer[rel + 16]) * 0.125f;
        float p = __expf(v - mnew);
        sum += p;
        Ps[tid * LD + c] = f2bf(p);
      }
      m_s[tid] = mnew;
      l_s[tid] = l_s[tid] * corr + sum;
      scale_s[tid] = corr;
    }
    __syncthreads();

    // O = O*corr + P @ V   (B column n = d -> Vts row d, K-dim = s)
    {
      const int hi = (lane >> 4) << 3;
      #pragma unroll
      for (int vr = 0; vr < 8; ++vr) {
        float c = scale_s[mt * 16 + vr + hi];
        o0[vr] *= c; o1[vr] *= c;
      }
      v16bf ap0 = load_a_frag(Ps, LD, mt * 16, 0, lane);
      v16bf ap1 = load_a_frag(Ps, LD, mt * 16, 32, lane);
      v16bf bv;
      bv = load_b_frag(Vts, LD, (nh * 2 + 0) * 16, 0,  lane); o0 = wmma_bf(ap0, bv, o0);
      bv = load_b_frag(Vts, LD, (nh * 2 + 0) * 16, 32, lane); o0 = wmma_bf(ap1, bv, o0);
      bv = load_b_frag(Vts, LD, (nh * 2 + 1) * 16, 0,  lane); o1 = wmma_bf(ap0, bv, o1);
      bv = load_b_frag(Vts, LD, (nh * 2 + 1) * 16, 32, lane); o1 = wmma_bf(ap1, bv, o1);
    }
    __syncthreads();
  }

  // finalize: O /= l, write ctx in (b, q, h*64+d) layout (bf16)
  const int hi = (lane >> 4) << 3, col = lane & 15;
  #pragma unroll
  for (int vr = 0; vr < 8; ++vr) {
    const int rloc = mt * 16 + vr + hi;
    const float inv = 1.f / l_s[rloc];
    const size_t base = headBase + (size_t)(q0 + rloc) * DMODEL;
    ctxb[base + (nh * 2 + 0) * 16 + col] = f2bf(o0[vr] * inv);
    ctxb[base + (nh * 2 + 1) * 16 + col] = f2bf(o1[vr] * inv);
  }
}

// ---------------------------------------------------------------------------
extern "C" void kernel_launch(void* const* d_in, const int* in_sizes, int n_in,
                              void* d_out, int out_size, void* d_ws, size_t ws_size,
                              hipStream_t stream) {
  const float* iQ  = (const float*)d_in[0];
  const float* iK  = (const float*)d_in[1];
  const float* iV  = (const float*)d_in[2];
  const float* Wq  = (const float*)d_in[3];
  const float* Wk  = (const float*)d_in[4];
  const float* Wv  = (const float*)d_in[5];
  const float* Wo  = (const float*)d_in[6];
  const float* rel = (const float*)d_in[7];

  const size_t NQ = (size_t)MROWS * DMODEL;    // 4096*1024 elements
  u16*   qbuf = (u16*)d_ws;
  u16*   kbuf = qbuf + NQ;
  u16*   vbuf = kbuf + NQ;
  u16*   ctxb = vbuf + NQ;
  float* qe   = (float*)(ctxb + NQ);           // 32 MiB offset, 16B aligned

  dim3 gg(DMODEL / 64, MROWS / 128);           // (16, 32)
  wmma_gemm_kernel<float, u16><<<gg, 256, 0, stream>>>(iQ, Wq, qbuf, MROWS, DMODEL, DMODEL);
  wmma_gemm_kernel<float, u16><<<gg, 256, 0, stream>>>(iK, Wk, kbuf, MROWS, DMODEL, DMODEL);
  wmma_gemm_kernel<float, u16><<<gg, 256, 0, stream>>>(iV, Wv, vbuf, MROWS, DMODEL, DMODEL);

  qe_kernel<<<BATCH * NHEAD * S_LEN, 64, 0, stream>>>(qbuf, rel, qe);

  attn_kernel<<<dim3(S_LEN / 64, BATCH * NHEAD), 256, 0, stream>>>(qbuf, kbuf, vbuf, qe, ctxb);

  wmma_gemm_kernel<u16, float><<<gg, 256, 0, stream>>>(ctxb, Wo, (float*)d_out, MROWS, DMODEL, DMODEL);
}
